// DeepseekV3Attention_55722905698686
// MI455X (gfx1250) — compile-verified, matrix operand order checked by reading, not run on hardware
//
#include <hip/hip_runtime.h>

// ---------------------------------------------------------------------------
// Types for CDNA5 WMMA (wave32, v_wmma_f32_16x16x32_bf16)
// ---------------------------------------------------------------------------
typedef __bf16 bf16_t;
typedef __attribute__((ext_vector_type(16))) __bf16 v16bf;
typedef __attribute__((ext_vector_type(8)))  __bf16 v8bf;
typedef __attribute__((ext_vector_type(8)))  float  v8f;

#define B_    2
#define S_    2048
#define D_    2048
#define H_    16
#define NOPE_ 128
#define ROPE_ 64
#define VDIM_ 128
#define QKD_  192
#define M_    (B_ * S_)          // 4096 token rows
#define SCALING_ 0.07216878364870323f   // 192^-0.5

// round-to-nearest-even fp32 -> bf16
__device__ __forceinline__ bf16_t f2bf(float f) {
    unsigned u = __builtin_bit_cast(unsigned, f);
    unsigned r = (u + 0x7FFFu + ((u >> 16) & 1u)) >> 16;
    unsigned short h = (unsigned short)r;
    return __builtin_bit_cast(bf16_t, h);
}

// ---------------------------------------------------------------------------
// Tensor Data Mover: one instruction stages an entire 2D bf16 tile
// global -> LDS (TENSORcnt, no VGPR round trip). D# per ISA 08 §8.
// Used double-buffered: issue tile k+1, then wait for tile k only
// (TENSORcnt completes in order within a wave).
// ---------------------------------------------------------------------------
#if __has_builtin(__builtin_amdgcn_tensor_load_to_lds)
#define HAVE_TDM 1
typedef unsigned u32x4 __attribute__((ext_vector_type(4)));
typedef int      i32x4 __attribute__((ext_vector_type(4)));
typedef int      i32x8 __attribute__((ext_vector_type(8)));

typedef __attribute__((address_space(3))) void lds_void_t;

__device__ __forceinline__ unsigned lds_offset(void* p) {
    return (unsigned)(unsigned long long)(lds_void_t*)p;
}

__device__ __forceinline__ void tdm_load_2d_bf16(unsigned lds_off, const void* gtile,
                                                 unsigned tile_rows, unsigned tile_cols,
                                                 unsigned row_stride_elems,
                                                 unsigned tensor_rows) {
    unsigned long long ga = (unsigned long long)(size_t)gtile;
    unsigned long long stride = (unsigned long long)row_stride_elems;
    unsigned td0 = row_stride_elems;   // tensor dim0 (elements per row)
    unsigned td1 = tensor_rows;        // tensor dim1 (rows)

    u32x4 g0;
    g0[0] = 1u;                                          // count=1, user descriptor
    g0[1] = lds_off;                                     // lds_addr (bytes)
    g0[2] = (unsigned)(ga & 0xFFFFFFFFull);              // global_addr[31:0]
    g0[3] = (unsigned)((ga >> 32) & 0x01FFFFFFull)       // global_addr[56:32]
            | (2u << 30);                                // type = 2 ("image")

    i32x8 g1;
    g1[0] = (int)(1u << 16);                             // data_size=1 (2B), mask=0
    g1[1] = (int)((td0 & 0xFFFFu) << 16);                // tensor_dim0[15:0]
    g1[2] = (int)(((td0 >> 16) & 0xFFFFu) | ((td1 & 0xFFFFu) << 16));
    g1[3] = (int)(((td1 >> 16) & 0xFFFFu) | ((tile_cols & 0xFFFFu) << 16)); // tile_dim0
    g1[4] = (int)(tile_rows & 0xFFFFu);                  // tile_dim1 (tile_dim2=0)
    g1[5] = (int)(unsigned)(stride & 0xFFFFFFFFull);     // tensor_dim0_stride[31:0]
    g1[6] = (int)(unsigned)((stride >> 32) & 0xFFFFull); // stride[47:32] (dim1_stride=0)
    g1[7] = 0;

    i32x4 z4 = {0, 0, 0, 0};
#if __clang_major__ >= 23
    i32x8 z8 = {0, 0, 0, 0, 0, 0, 0, 0};
    __builtin_amdgcn_tensor_load_to_lds(g0, g1, z4, z4, z8, 0);
#else
    __builtin_amdgcn_tensor_load_to_lds(g0, g1, z4, z4, 0);
#endif
}

template <int N>
__device__ __forceinline__ void tdm_wait() {
#if __has_builtin(__builtin_amdgcn_s_wait_tensorcnt)
    __builtin_amdgcn_s_wait_tensorcnt(N);
#else
    asm volatile("s_wait_tensorcnt %0" ::"i"(N) : "memory");
#endif
}
#endif // HAVE_TDM

// ---------------------------------------------------------------------------
// WMMA fragment loader (ISA 7.12.2): 16 rows x 32 K, K-major storage.
// ---------------------------------------------------------------------------
__device__ __forceinline__ v16bf load_frag_bf16(const bf16_t* base, int row0,
                                                int k0, int stride) {
    const int lane = threadIdx.x & 31;
    const int r    = row0 + (lane & 15);
    const int kb   = (lane >> 4) << 3;          // 0 or 8
    const bf16_t* p = base + (size_t)r * stride + k0 + kb;
    v16bf f;
    ((v8bf*)&f)[0] = *(const v8bf*)(p);         // K = kb .. kb+7
    ((v8bf*)&f)[1] = *(const v8bf*)(p + 16);    // K = 16+kb .. 16+kb+7
    return f;
}

// ---------------------------------------------------------------------------
// fp32 -> bf16 conversion
// ---------------------------------------------------------------------------
__global__ __launch_bounds__(256) void cvt_f32_bf16(const float* __restrict__ src,
                                                    bf16_t* __restrict__ dst, int n) {
    int i = blockIdx.x * 256 + threadIdx.x;
    if (i < n) dst[i] = f2bf(src[i]);
}

// ---------------------------------------------------------------------------
// C[M,N] (fp32) = A[M,K] (bf16 row-major) @ W[N,K]^T (bf16 row-major)
// 256 threads = 8 waves; block tile 128x64; wave tile 32x32; K step 64.
// TDM double-buffered: tile k+1 streams into the other LDS bank while
// WMMAs consume tile k.
// ---------------------------------------------------------------------------
#define GEMM_BM 128
#define GEMM_BN 64
#define GEMM_BK 64

__global__ __launch_bounds__(256) void gemm_bf16_nt(const bf16_t* __restrict__ A,
                                                    const bf16_t* __restrict__ W,
                                                    float* __restrict__ C,
                                                    int M, int N, int K) {
#if defined(HAVE_TDM)
    __shared__ __align__(16) bf16_t As[2][GEMM_BM][GEMM_BK];   // 2 x 16 KB
    __shared__ __align__(16) bf16_t Ws[2][GEMM_BN][GEMM_BK];   // 2 x  8 KB
#else
    __shared__ __align__(16) bf16_t As[1][GEMM_BM][GEMM_BK];
    __shared__ __align__(16) bf16_t Ws[1][GEMM_BN][GEMM_BK];
#endif

    const int tid  = threadIdx.x;
    const int wave = tid >> 5;
    const int lane = tid & 31;
    const int lane16 = lane & 15;
    const int hi     = lane >> 4;

    const int m_block = blockIdx.y * GEMM_BM;
    const int n_block = blockIdx.x * GEMM_BN;
    const int wm = (wave >> 1) * 32;   // 0,32,64,96
    const int wn = (wave & 1) * 32;    // 0,32

    v8f acc[2][2] = {};
    const int nk = K / GEMM_BK;

#if defined(HAVE_TDM)
    if (tid == 0) {   // prologue: tile 0 -> bank 0 (TDM ignores EXEC; one wave issues)
        tdm_load_2d_bf16(lds_offset(&As[0][0][0]), A + (size_t)m_block * K,
                         GEMM_BM, GEMM_BK, K, M);
        tdm_load_2d_bf16(lds_offset(&Ws[0][0][0]), W + (size_t)n_block * K,
                         GEMM_BN, GEMM_BK, K, N);
    }
#endif

    for (int kt = 0; kt < nk; kt++) {
#if defined(HAVE_TDM)
        const int buf  = kt & 1;
        const bool more = (kt + 1) < nk;
        if (more && tid == 0) {       // stream next tile into the other bank
            int k1 = (kt + 1) * GEMM_BK;
            tdm_load_2d_bf16(lds_offset(&As[buf ^ 1][0][0]),
                             A + (size_t)m_block * K + k1, GEMM_BM, GEMM_BK, K, M);
            tdm_load_2d_bf16(lds_offset(&Ws[buf ^ 1][0][0]),
                             W + (size_t)n_block * K + k1, GEMM_BN, GEMM_BK, K, N);
        }
        if (more) tdm_wait<2>();      // drain current tile only; next stays in flight
        else      tdm_wait<0>();
        __syncthreads();
#else
        const int buf = 0;
        const int k0  = kt * GEMM_BK;
        {   // two-phase: all global loads first, then LDS stores
            v8bf t[6];
#pragma unroll
            for (int u = 0; u < 4; u++) {
                int i = tid + u * 256, r = i >> 3, c = (i & 7) * 8;
                t[u] = *(const v8bf*)(A + (size_t)(m_block + r) * K + k0 + c);
            }
#pragma unroll
            for (int u = 0; u < 2; u++) {
                int i = tid + u * 256, r = i >> 3, c = (i & 7) * 8;
                t[4 + u] = *(const v8bf*)(W + (size_t)(n_block + r) * K + k0 + c);
            }
#pragma unroll
            for (int u = 0; u < 4; u++) {
                int i = tid + u * 256, r = i >> 3, c = (i & 7) * 8;
                *(v8bf*)&As[0][r][c] = t[u];
            }
#pragma unroll
            for (int u = 0; u < 2; u++) {
                int i = tid + u * 256, r = i >> 3, c = (i & 7) * 8;
                *(v8bf*)&Ws[0][r][c] = t[4 + u];
            }
        }
        __syncthreads();
#endif

        const bf16_t* Ab = &As[buf][0][0];
        const bf16_t* Wb = &Ws[buf][0][0];
#pragma unroll
        for (int c = 0; c < 2; c++)
#pragma unroll
            for (int i = 0; i < 2; i++) {
                v16bf af = load_frag_bf16(Ab, wm + i * 16, c * 32, GEMM_BK);
#pragma unroll
                for (int jj = 0; jj < 2; jj++) {
                    v16bf bf = load_frag_bf16(Wb, wn + jj * 16, c * 32, GEMM_BK);
                    acc[i][jj] = __builtin_amdgcn_wmma_f32_16x16x32_bf16(
                        false, af, false, bf, (short)0, acc[i][jj], false, false);
                }
            }
        __syncthreads();   // all waves done with this bank before it is refilled
    }

    // C layout: VGPR e -> M = e + 8*hi, N = lane&15
#pragma unroll
    for (int i = 0; i < 2; i++)
#pragma unroll
        for (int j = 0; j < 2; j++)
#pragma unroll
            for (int e = 0; e < 8; e++) {
                int m = m_block + wm + i * 16 + hi * 8 + e;
                int n = n_block + wn + j * 16 + lane16;
                C[(size_t)m * N + n] = acc[i][j][e];
            }
}

// ---------------------------------------------------------------------------
// RMSNorm: one block (256 threads = 8 waves) per row; fp32 in, bf16 out.
// ---------------------------------------------------------------------------
__global__ __launch_bounds__(256) void rmsnorm_kernel(const float* __restrict__ src,
                                                      int srcStride, int cols,
                                                      const float* __restrict__ w,
                                                      bf16_t* __restrict__ dst,
                                                      int dstStride) {
    const int row = blockIdx.x;
    const float* x = src + (size_t)row * srcStride;
    float ss = 0.f;
    for (int c = threadIdx.x; c < cols; c += 256) { float v = x[c]; ss += v * v; }
    for (int off = 16; off > 0; off >>= 1) ss += __shfl_xor(ss, off, 32);
    __shared__ float red[8];
    __shared__ float stot;
    if ((threadIdx.x & 31) == 0) red[threadIdx.x >> 5] = ss;
    __syncthreads();
    if (threadIdx.x == 0) {
        float t = 0.f;
        for (int i = 0; i < 8; i++) t += red[i];
        stot = t;
    }
    __syncthreads();
    const float rstd = rsqrtf(stot / (float)cols + 1e-6f);
    bf16_t* dr = dst + (size_t)row * dstStride;
    for (int c = threadIdx.x; c < cols; c += 256) dr[c] = f2bf(x[c] * rstd * w[c]);
}

// ---------------------------------------------------------------------------
// Assemble Q/K/V (bf16, (B,H,S,d) layout). RoPE on q[...,128:] / ckv[...,512:];
// softmax scaling folded into Q. One block per token (b,s).
// ---------------------------------------------------------------------------
__global__ __launch_bounds__(256) void assemble_qkv(const float* __restrict__ q,
                                                    const float* __restrict__ ckv,
                                                    const float* __restrict__ kv,
                                                    const float* __restrict__ cosb,
                                                    const float* __restrict__ sinb,
                                                    bf16_t* __restrict__ Qb,
                                                    bf16_t* __restrict__ Kb,
                                                    bf16_t* __restrict__ Vb,
                                                    int S) {
    const int bs = blockIdx.x;            // b*S + s
    const int b  = bs / S;
    const int s  = bs - b * S;
    const float* qr = q   + (size_t)bs * (H_ * QKD_);
    const float* ck = ckv + (size_t)bs * 576;
    const float* kr = kv  + (size_t)bs * (H_ * 256);
    const float* cs = cosb + (size_t)bs * ROPE_;
    const float* sn = sinb + (size_t)bs * ROPE_;

    for (int t = threadIdx.x; t < H_ * QKD_; t += 256) {
        int h = t / QKD_, d = t - h * QKD_;
        float v;
        if (d < NOPE_) {
            v = qr[h * QKD_ + d];
        } else {
            int j = d - NOPE_;
            float x  = qr[h * QKD_ + NOPE_ + j];
            float rx = (j < 32) ? -qr[h * QKD_ + NOPE_ + j + 32]
                                :  qr[h * QKD_ + NOPE_ + j - 32];
            v = x * cs[j] + rx * sn[j];
        }
        Qb[(((size_t)(b * H_ + h)) * S + s) * QKD_ + d] = f2bf(v * SCALING_);
    }
    for (int t = threadIdx.x; t < H_ * QKD_; t += 256) {
        int h = t / QKD_, d = t - h * QKD_;
        float v;
        if (d < NOPE_) {
            v = kr[h * 256 + d];
        } else {
            int j = d - NOPE_;
            float x  = ck[512 + j];
            float rx = (j < 32) ? -ck[512 + j + 32] : ck[512 + j - 32];
            v = x * cs[j] + rx * sn[j];
        }
        Kb[(((size_t)(b * H_ + h)) * S + s) * QKD_ + d] = f2bf(v);
    }
    for (int t = threadIdx.x; t < H_ * VDIM_; t += 256) {
        int h = t >> 7, d = t & 127;
        Vb[(((size_t)(b * H_ + h)) * S + s) * VDIM_ + d] = f2bf(kr[h * 256 + NOPE_ + d]);
    }
}

// ---------------------------------------------------------------------------
// Flash attention: 128 threads (4 waves); one (b,h, 64-query tile) per block;
// each wave owns 16 query rows. K tiles double-buffered via TDM (next tile in
// flight under the QK^T WMMAs); V transposed cooperatively; P tile overlaid
// on the dead half of the current K bank (barriers fence the reuse).
// ---------------------------------------------------------------------------
__global__ __launch_bounds__(128) void flash_attn(const bf16_t* __restrict__ Qb,
                                                  const bf16_t* __restrict__ Kb,
                                                  const bf16_t* __restrict__ Vb,
                                                  bf16_t* __restrict__ Ob,
                                                  int S) {
    const int qt = blockIdx.x;            // query tile (64 rows)
    const int bh = blockIdx.y;            // b*H + h
    const int b  = bh >> 4;
    const int h  = bh & 15;

    const bf16_t* Q  = Qb + (size_t)bh * S * QKD_;
    const bf16_t* Kp = Kb + (size_t)bh * S * QKD_;
    const bf16_t* Vp = Vb + (size_t)bh * S * VDIM_;

    const int tid  = threadIdx.x;
    const int wave = tid >> 5;
    const int lane = tid & 31;
    const int lane16 = lane & 15;
    const int hi     = lane >> 4;

#if defined(HAVE_TDM)
    __shared__ __align__(16) bf16_t Ks[2][64][QKD_];  // 48 KB (double-buffered)
#else
    __shared__ __align__(16) bf16_t Ks[1][64][QKD_];  // 24 KB
    __shared__ __align__(16) bf16_t PsArr[4][16][64]; //  8 KB
#endif
    __shared__ __align__(16) bf16_t Vts[VDIM_][64];   // 16 KB (V transposed)

    const int qrow0 = qt * 64 + wave * 16;
    v16bf qf[6];
#pragma unroll
    for (int c = 0; c < 6; c++) qf[c] = load_frag_bf16(Q, qrow0, c * 32, QKD_);

    v8f o_acc[8] = {};
    float m_r[8], l_r[8];
#pragma unroll
    for (int e = 0; e < 8; e++) { m_r[e] = -1e30f; l_r[e] = 0.f; }

    const int ktiles = qt + 1;            // causal

#if defined(HAVE_TDM)
    if (tid == 0)                         // prologue: K tile 0 -> bank 0
        tdm_load_2d_bf16(lds_offset(&Ks[0][0][0]), Kp, 64, QKD_, QKD_, S);
#endif

    for (int j = 0; j < ktiles; j++) {
#if defined(HAVE_TDM)
        const int buf = j & 1;
        const bool more = (j + 1) < ktiles;
        if (more && tid == 0)             // next K tile streams under this tile's math
            tdm_load_2d_bf16(lds_offset(&Ks[buf ^ 1][0][0]),
                             Kp + (size_t)(j + 1) * 64 * QKD_, 64, QKD_, QKD_, S);
#else
        const int buf = 0;
        for (int i = tid; i < 64 * 24; i += 128) {
            int r = i / 24, c = (i % 24) * 8;
            *(v8bf*)&Ks[0][r][c] = *(const v8bf*)(Kp + (size_t)(j * 64 + r) * QKD_ + c);
        }
#endif
        // V tile, transposed into LDS (two-phase batches) — overlaps the TDM
        for (int base = 0; base < 64 * 16; base += 128 * 4) {
            v8bf tmp[4];
#pragma unroll
            for (int u = 0; u < 4; u++) {
                int i = base + tid + u * 128;
                int r = i >> 4, c0 = (i & 15) * 8;
                tmp[u] = *(const v8bf*)(Vp + (size_t)(j * 64 + r) * VDIM_ + c0);
            }
#pragma unroll
            for (int u = 0; u < 4; u++) {
                int i = base + tid + u * 128;
                int r = i >> 4, c0 = (i & 15) * 8;
#pragma unroll
                for (int e = 0; e < 8; e++) Vts[c0 + e][r] = tmp[u][e];
            }
        }
#if defined(HAVE_TDM)
        if (more) tdm_wait<1>();          // current K tile done; next stays in flight
        else      tdm_wait<0>();
#endif
        __syncthreads();                  // (a) K + V visible to all waves

        // scores: 16 query rows x 64 keys, contraction over d=192 (6 chunks)
        v8f sc[4] = {};
#pragma unroll
        for (int n = 0; n < 4; n++) {
#pragma unroll
            for (int c = 0; c < 6; c++) {
                v16bf kf = load_frag_bf16(&Ks[buf][0][0], n * 16, c * 32, QKD_);
                sc[n] = __builtin_amdgcn_wmma_f32_16x16x32_bf16(
                    false, qf[c], false, kf, (short)0, sc[n], false, false);
            }
        }
        __syncthreads();                  // (b) all waves done reading Ks[buf]:
                                          //     its space can now host the P tiles

        // causal mask: row = qrow0 + e + 8*hi, col = j*64 + n*16 + lane16
#pragma unroll
        for (int n = 0; n < 4; n++)
#pragma unroll
            for (int e = 0; e < 8; e++) {
                int grow = qrow0 + e + hi * 8;
                int gcol = j * 64 + n * 16 + lane16;
                if (gcol > grow) sc[n][e] = -1e30f;
            }

        // row max across the 16 lanes holding each row
        float mnew[8];
#pragma unroll
        for (int e = 0; e < 8; e++) {
            float v = fmaxf(fmaxf(sc[0][e], sc[1][e]), fmaxf(sc[2][e], sc[3][e]));
            for (int off = 8; off > 0; off >>= 1) v = fmaxf(v, __shfl_xor(v, off, 16));
            mnew[e] = fmaxf(m_r[e], v);
        }

        float scale_o[8];
#pragma unroll
        for (int e = 0; e < 8; e++) {
            scale_o[e] = __expf(m_r[e] - mnew[e]);
            m_r[e] = mnew[e];
            l_r[e] *= scale_o[e];
        }
#pragma unroll
        for (int f = 0; f < 8; f++)
#pragma unroll
            for (int e = 0; e < 8; e++) o_acc[f][e] *= scale_o[e];

        // P = exp(S - m); row sums into l
#pragma unroll
        for (int n = 0; n < 4; n++)
#pragma unroll
            for (int e = 0; e < 8; e++) sc[n][e] = __expf(sc[n][e] - m_r[e]);
#pragma unroll
        for (int e = 0; e < 8; e++) {
            float ps = sc[0][e] + sc[1][e] + sc[2][e] + sc[3][e];
            for (int off = 8; off > 0; off >>= 1) ps += __shfl_xor(ps, off, 16);
            l_r[e] += ps;
        }

        // write P (C layout) into per-wave LDS row-major [16][64]
#if defined(HAVE_TDM)
        bf16_t* Psw = &Ks[buf][0][0] + wave * (16 * 64);   // overlay on dead bank
#else
        bf16_t* Psw = &PsArr[wave][0][0];
#endif
#pragma unroll
        for (int n = 0; n < 4; n++)
#pragma unroll
            for (int e = 0; e < 8; e++)
                Psw[(e + hi * 8) * 64 + n * 16 + lane16] = f2bf(sc[n][e]);
        __builtin_amdgcn_wave_barrier();  // wave-local DS ordering (write -> frag read)

        // O += P(16x64) @ V(64x128)
#pragma unroll
        for (int kk = 0; kk < 2; kk++) {
            v16bf pf = load_frag_bf16(Psw, 0, kk * 32, 64);
#pragma unroll
            for (int f = 0; f < 8; f++) {
                v16bf vf = load_frag_bf16(&Vts[0][0], f * 16, kk * 32, 64);
                o_acc[f] = __builtin_amdgcn_wmma_f32_16x16x32_bf16(
                    false, pf, false, vf, (short)0, o_acc[f], false, false);
            }
        }
        __syncthreads();                  // (c) Vts / Ks[buf] free for next iteration
    }

    // epilogue: O /= l, store bf16 at (b*S + row, h*128 + d)
#pragma unroll
    for (int e = 0; e < 8; e++) l_r[e] = 1.0f / l_r[e];
#pragma unroll
    for (int f = 0; f < 8; f++)
#pragma unroll
        for (int e = 0; e < 8; e++) {
            int srow = qrow0 + e + hi * 8;
            int d    = f * 16 + lane16;
            Ob[((size_t)(b * S + srow) * H_ + h) * VDIM_ + d] =
                f2bf(o_acc[f][e] * l_r[e]);
        }
}

// ---------------------------------------------------------------------------
// Host-side launch
// ---------------------------------------------------------------------------
extern "C" void kernel_launch(void* const* d_in, const int* in_sizes, int n_in,
                              void* d_out, int out_size, void* d_ws, size_t ws_size,
                              hipStream_t stream) {
    (void)in_sizes; (void)n_in; (void)out_size; (void)ws_size;

    const float* hidden  = (const float*)d_in[0];
    const float* cosb    = (const float*)d_in[1];
    const float* sinb    = (const float*)d_in[2];
    const float* q_a_W   = (const float*)d_in[3];
    const float* q_a_nw  = (const float*)d_in[4];
    const float* q_b_W   = (const float*)d_in[5];
    const float* kv_a_W  = (const float*)d_in[6];
    const float* kv_a_nw = (const float*)d_in[7];
    const float* kv_b_W  = (const float*)d_in[8];
    const float* o_W     = (const float*)d_in[9];
    float* out = (float*)d_out;

    char* wsp = (char*)d_ws;
    auto take = [&](size_t bytes) -> void* {
        void* p = (void*)wsp;
        wsp += (bytes + 255) & ~(size_t)255;
        return p;
    };

    bf16_t* hbf    = (bf16_t*)take((size_t)M_ * D_ * 2);
    bf16_t* wqa    = (bf16_t*)take((size_t)1536 * D_ * 2);
    bf16_t* wqb    = (bf16_t*)take((size_t)(H_ * QKD_) * 1536 * 2);
    bf16_t* wkva   = (bf16_t*)take((size_t)576 * D_ * 2);
    bf16_t* wkvb   = (bf16_t*)take((size_t)(H_ * 256) * 512 * 2);
    bf16_t* wo     = (bf16_t*)take((size_t)D_ * (H_ * VDIM_) * 2);
    float*  qlat   = (float*) take((size_t)M_ * 1536 * 4);
    bf16_t* qnorm  = (bf16_t*)take((size_t)M_ * 1536 * 2);
    float*  qbuf   = (float*) take((size_t)M_ * (H_ * QKD_) * 4);
    float*  ckv    = (float*) take((size_t)M_ * 576 * 4);
    bf16_t* kvnorm = (bf16_t*)take((size_t)M_ * 512 * 2);
    float*  kvbuf  = (float*) take((size_t)M_ * (H_ * 256) * 4);
    bf16_t* Qb     = (bf16_t*)take((size_t)B_ * H_ * S_ * QKD_ * 2);
    bf16_t* Kb     = (bf16_t*)take((size_t)B_ * H_ * S_ * QKD_ * 2);
    bf16_t* Vb     = (bf16_t*)take((size_t)B_ * H_ * S_ * VDIM_ * 2);
    bf16_t* attn   = (bf16_t*)take((size_t)M_ * (H_ * VDIM_) * 2);

    auto cvt = [&](const float* s, bf16_t* dd, int n) {
        cvt_f32_bf16<<<(n + 255) / 256, 256, 0, stream>>>(s, dd, n);
    };
    cvt(hidden, hbf,  M_ * D_);
    cvt(q_a_W,  wqa,  1536 * D_);
    cvt(q_b_W,  wqb,  H_ * QKD_ * 1536);
    cvt(kv_a_W, wkva, 576 * D_);
    cvt(kv_b_W, wkvb, H_ * 256 * 512);
    cvt(o_W,    wo,   D_ * H_ * VDIM_);

    auto gemm = [&](const bf16_t* A, const bf16_t* W, float* C, int M, int N, int K) {
        gemm_bf16_nt<<<dim3(N / GEMM_BN, M / GEMM_BM), 256, 0, stream>>>(A, W, C, M, N, K);
    };

    gemm(hbf, wqa,  qlat, M_, 1536, D_);                 // q_a projection
    gemm(hbf, wkva, ckv,  M_, 576,  D_);                 // kv_a projection
    rmsnorm_kernel<<<M_, 256, 0, stream>>>(qlat, 1536, 1536, q_a_nw,  qnorm,  1536);
    rmsnorm_kernel<<<M_, 256, 0, stream>>>(ckv,  576,  512,  kv_a_nw, kvnorm, 512);
    gemm(qnorm,  wqb,  qbuf,  M_, H_ * QKD_, 1536);      // q_b projection
    gemm(kvnorm, wkvb, kvbuf, M_, H_ * 256,  512);       // kv_b projection

    assemble_qkv<<<M_, 256, 0, stream>>>(qbuf, ckv, kvbuf, cosb, sinb, Qb, Kb, Vb, S_);

    flash_attn<<<dim3(S_ / 64, B_ * H_), 128, 0, stream>>>(Qb, Kb, Vb, attn, S_);

    gemm(attn, wo, out, M_, D_, H_ * VDIM_);             // output projection
}
